// MultiHeadedAttention_18983755448762
// MI455X (gfx1250) — matrix operand order repaired and checked by reading, not compiled, not run on hardware
//
#include <hip/hip_runtime.h>
#include <cstdint>
#include <cstddef>

// ---------------------------------------------------------------------------
// MultiHeadedAttention (linear attention) for MI455X / gfx1250, wave32 + WMMA.
//   B=8, D_MODEL=256, HEADS=8, HEAD_DIM=32, L=4096 (L derived from in_sizes).
// All matrix math on v_wmma_f32_16x16x32_bf16; accumulation/epilogues in fp32.
// ---------------------------------------------------------------------------

#define DM    256
#define NH    8
#define BATCH 8
#define EPSV  1e-6f

typedef __bf16   v16bf __attribute__((ext_vector_type(16)));
typedef float    v8f   __attribute__((ext_vector_type(8)));
typedef uint32_t u32x4 __attribute__((ext_vector_type(4)));

struct FragBits { u32x4 lo, hi; };   // 32 bytes == 16 x bf16

static __device__ __forceinline__ v16bf make_frag(u32x4 lo, u32x4 hi) {
  FragBits fb{lo, hi};
  return __builtin_bit_cast(v16bf, fb);
}

static __device__ __forceinline__ uint32_t pack2_bf16(float a, float b) {
  unsigned short ha = __builtin_bit_cast(unsigned short, (__bf16)a);
  unsigned short hb = __builtin_bit_cast(unsigned short, (__bf16)b);
  return (uint32_t)ha | ((uint32_t)hb << 16);
}

static __device__ __forceinline__ v8f wmma_bf16(v16bf a, v16bf b, v8f c) {
  // 8-arg form: (neg_a, A, neg_b, B, c_mod, C, reuse_a, reuse_b)
  return __builtin_amdgcn_wmma_f32_16x16x32_bf16(false, a, false, b,
                                                 (short)0, c, false, false);
}

static __device__ __forceinline__ v8f zero_v8f() {
  v8f z = {0.f, 0.f, 0.f, 0.f, 0.f, 0.f, 0.f, 0.f};
  return z;
}

// ---------------------------------------------------------------------------
// Kernel 1/4: O_b[256,L] = W[256,256] * X_b[256,L] + bias, with epilogue.
//   MODE 0: none    MODE 1: elu(y)+1 = (y>0 ? y+1 : exp(y))    MODE 2: y*scale
// Tile: 128(M) x 128(N) per block, KT=32, 256 threads = 8 wave32.
// Double-buffered LDS; global loads for step k+1 are register-staged and
// issued before the WMMAs of step k (latency hides under the matrix pipe).
// LDS holds bf16 pairs packed as u32 in natural k order (u32 kp = k 2kp,2kp+1); rows padded to 20
// u32 so fragment reads are 16B-aligned ds_load_b128 with spread banks.
// ---------------------------------------------------------------------------
template <int MODE>
__global__ __launch_bounds__(256)
void proj_gemm_kernel(const float* __restrict__ X, const float* __restrict__ W,
                      const float* __restrict__ bias, float* __restrict__ O,
                      int L, float scale) {
  __shared__ __align__(16) uint32_t Asm[2][128 * 20];  // [buf][m][kpair]
  __shared__ __align__(16) uint32_t Bsm[2][128 * 20];  // [buf][n][kpair]

  const int tid  = threadIdx.x;
  const int wv   = tid >> 5;          // wave 0..7  -> M strip 16*wv
  const int lane = tid & 31;
  const int l16  = lane & 15;
  const int hs   = lane >> 4;         // lane half-select
  const int nTile = blockIdx.x * 128;
  const int mTile = blockIdx.y * 128;
  const int b     = blockIdx.z;

  const float* Xb = X + (size_t)b * DM * L;
  float*       Ob = O + (size_t)b * DM * L;

  // per-thread staging assignments (fixed across k-steps)
  const int aM  = tid >> 4,  aKp = tid & 15;    // A: thread covers (aM + 16*it? no)
  // A: linear index i = tid + 256*it -> m = i>>4, kp = i&15.
  // B: linear index i = tid + 256*it -> kp = i>>7, n = i&127.

  v8f acc[8];
#pragma unroll
  for (int i = 0; i < 8; ++i) acc[i] = zero_v8f();

  float2 aReg[8];
  float  bReg0[8], bReg1[8];

  auto load_regs = [&](int k0) {
#pragma unroll
    for (int it = 0; it < 8; ++it) {            // W slab: 128 x 32
      const int i = tid + it * 256;
      const int m = i >> 4, kp = i & 15;
      aReg[it] = *reinterpret_cast<const float2*>(
          W + (size_t)(mTile + m) * DM + k0 + 2 * kp);
    }
#pragma unroll
    for (int it = 0; it < 8; ++it) {            // X slab: 32 x 128
      const int i = tid + it * 256;
      const int kp = i >> 7, n = i & 127;
      const float* xp = Xb + (size_t)(k0 + 2 * kp) * L + nTile + n;
      bReg0[it] = xp[0];
      bReg1[it] = xp[L];
    }
  };

  auto store_lds = [&](int buf) {
#pragma unroll
    for (int it = 0; it < 8; ++it) {
      const int i = tid + it * 256;
      const int m = i >> 4, kp = i & 15;
      Asm[buf][m * 20 + kp] = pack2_bf16(aReg[it].x, aReg[it].y);
    }
#pragma unroll
    for (int it = 0; it < 8; ++it) {
      const int i = tid + it * 256;
      const int kp = i >> 7, n = i & 127;
      Bsm[buf][n * 20 + kp] = pack2_bf16(bReg0[it], bReg1[it]);
    }
  };

  auto compute = [&](int buf) {
    // A fragment (16x32, this wave's M strip): two ds_load_b128
    const int arow = (wv * 16 + l16) * 20;
    u32x4 alo = *reinterpret_cast<const u32x4*>(&Asm[buf][arow + hs * 4]);
    u32x4 ahi = *reinterpret_cast<const u32x4*>(&Asm[buf][arow + 8 + hs * 4]);
    v16bf af = make_frag(alo, ahi);
#pragma unroll
    for (int nt = 0; nt < 8; ++nt) {
      const int brow = (nt * 16 + l16) * 20 + hs * 4;
      u32x4 blo = *reinterpret_cast<const u32x4*>(&Bsm[buf][brow]);
      u32x4 bhi = *reinterpret_cast<const u32x4*>(&Bsm[buf][brow + 8]);
      acc[nt] = wmma_bf16(af, make_frag(blo, bhi), acc[nt]);
    }
  };

  load_regs(0);
  store_lds(0);
  __syncthreads();

#pragma unroll
  for (int ks = 0; ks < 8; ++ks) {              // K = 256 in 8 slabs of 32
    if (ks + 1 < 8) {
      load_regs((ks + 1) * 32);                 // in flight during compute
      if (ks + 2 < 8) {                         // HBM->L2 two slabs ahead
        const float* pf =
            Xb + (size_t)((ks + 2) * 32 + (tid >> 3)) * L + nTile + (tid & 7) * 16;
        __builtin_prefetch(pf, 0, 1);
      }
    }
    compute(ks & 1);
    if (ks + 1 < 8) {
      store_lds((ks + 1) & 1);                  // other parity: no race w/ readers
      __syncthreads();
    }
  }

  // epilogue + store (C layout: m = r + 8*hs, n = l16)
#pragma unroll
  for (int nt = 0; nt < 8; ++nt) {
    const int col = nTile + nt * 16 + l16;
#pragma unroll
    for (int r = 0; r < 8; ++r) {
      const int o = mTile + wv * 16 + r + hs * 8;
      float y = acc[nt][r] + bias[o];
      if (MODE == 1)      y = (y > 0.f) ? (y + 1.f) : __expf(y);
      else if (MODE == 2) y *= scale;
      Ob[(size_t)o * L + col] = y;
    }
  }
  (void)aM; (void)aKp;
}

// ---------------------------------------------------------------------------
// Kernel 2: per (b,h):  KV[q,d] = sum_m V[q,m]*K[d,m]  (32x32, K-dim = L)
//           ksum[d]    = sum_m K[d,m]   (fp32, from pre-rounding loads)
// One block per (b,h); 8 waves each reduce L/8 columns; LDS tree reduction.
// Head h's row d lives at channel c = d*8 + h  (row stride 8*L floats).
// ---------------------------------------------------------------------------
__global__ __launch_bounds__(256)
void kv_ksum_kernel(const float* __restrict__ Kb, const float* __restrict__ Vb,
                    float* __restrict__ KV, float* __restrict__ KS, int L) {
  __shared__ __align__(16) float red[8][1024];
  __shared__ float kred[8][64];

  const int bh = blockIdx.x, b = bh >> 3, h = bh & 7;
  const int tid = threadIdx.x, wv = tid >> 5, lane = tid & 31;
  const int l16 = lane & 15, hs = lane >> 4;
  const float* Kp = Kb + (size_t)b * DM * L;
  const float* Vp = Vb + (size_t)b * DM * L;

  v8f acc[2][2];
#pragma unroll
  for (int i = 0; i < 2; ++i)
#pragma unroll
    for (int j = 0; j < 2; ++j) acc[i][j] = zero_v8f();
  float ks[2] = {0.f, 0.f};

  const int chunk = L >> 3;          // 512 columns per wave
  const int c0 = wv * chunk;
  for (int s = 0; s < chunk; s += 32) {
    const int kk0 = c0 + s;
    v16bf aF[2], bF[2];
#pragma unroll
    for (int mt = 0; mt < 2; ++mt) {   // A = V rows (q), 16x32 layout
      const int q = mt * 16 + l16;
      const float* vrow = Vp + (size_t)(q * NH + h) * L + kk0;
#pragma unroll
      for (int j = 0; j < 8; ++j) {
        const int cc = hs * 16 + j * 2;
        const float2 v2 = *reinterpret_cast<const float2*>(vrow + cc);
        aF[mt][2 * j]     = (__bf16)v2.x;
        aF[mt][2 * j + 1] = (__bf16)v2.y;
      }
    }
#pragma unroll
    for (int nt = 0; nt < 2; ++nt) {   // B = K^T: B[kk][n] = K[d=n][m=kk]
      const int d = nt * 16 + l16;
      const float* krow = Kp + (size_t)(d * NH + h) * L + kk0 + hs * 16;
#pragma unroll
      for (int j = 0; j < 8; ++j) {
        const float2 k2 = *reinterpret_cast<const float2*>(krow + 2 * j);
        bF[nt][2 * j]     = (__bf16)k2.x;
        bF[nt][2 * j + 1] = (__bf16)k2.y;
        ks[nt] += k2.x + k2.y;         // fp32 ksum partial
      }
    }
#pragma unroll
    for (int mt = 0; mt < 2; ++mt)
#pragma unroll
      for (int nt = 0; nt < 2; ++nt)
        acc[mt][nt] = wmma_bf16(aF[mt], bF[nt], acc[mt][nt]);
  }

  // deterministic cross-wave reduction via LDS
#pragma unroll
  for (int mt = 0; mt < 2; ++mt)
#pragma unroll
    for (int nt = 0; nt < 2; ++nt)
#pragma unroll
      for (int r = 0; r < 8; ++r)
        red[wv][(mt * 16 + r + hs * 8) * 32 + nt * 16 + l16] = acc[mt][nt][r];
  kred[wv][lane]      = ks[0];   // d = l16      (two contributors: lane, lane^16)
  kred[wv][32 + lane] = ks[1];   // d = 16 + l16
  __syncthreads();

  for (int o = tid; o < 1024; o += 256) {
    float s = 0.f;
#pragma unroll
    for (int w = 0; w < 8; ++w) s += red[w][o];
    KV[(size_t)bh * 1024 + o] = s;
  }
  if (tid < 32) {
    const int d = tid, nt = d >> 4, tl = d & 15;
    float s = 0.f;
#pragma unroll
    for (int w = 0; w < 8; ++w)
      s += kred[w][nt * 32 + tl] + kred[w][nt * 32 + tl + 16];
    KS[bh * 32 + d] = s;
  }
}

// ---------------------------------------------------------------------------
// Kernel 3: per (b,h,col):  out[q] = (KV[q,:] . Q[:,col]) * L/(Q.ksum + eps)
// K-dim = 32 -> exactly one WMMA per 16x16 tile. Denominator in fp32 from the
// same loads, combined across lane halves with shfl_xor(16). Heads merged on
// store: channel c = q*8 + h.
// ---------------------------------------------------------------------------
__global__ __launch_bounds__(256)
void attn_out_kernel(const float* __restrict__ Qb, const float* __restrict__ KV,
                     const float* __restrict__ KS, float* __restrict__ X2, int L) {
  const int tid = threadIdx.x, wv = tid >> 5, lane = tid & 31;
  const int l16 = lane & 15, hs = lane >> 4;
  const int h = blockIdx.y, b = blockIdx.z;
  const int col = blockIdx.x * 128 + wv * 16 + l16;

  const float* Qp  = Qb + (size_t)b * DM * L;
  const float* kv  = KV + (size_t)(b * NH + h) * 1024;
  const float* ksv = KS + (size_t)(b * NH + h) * 32;

  // A fragments from KV (fp32 -> bf16), 16x32 layout, K-dim = d
  v16bf aF[2];
#pragma unroll
  for (int mt = 0; mt < 2; ++mt) {
    const int q = mt * 16 + l16;
#pragma unroll
    for (int j = 0; j < 8; ++j) {
      const int d0 = hs * 16 + 2 * j;
      const float2 t2 = *reinterpret_cast<const float2*>(kv + q * 32 + d0);
      aF[mt][2 * j]     = (__bf16)t2.x;
      aF[mt][2 * j + 1] = (__bf16)t2.y;
    }
  }

  // B fragment from Q (column `col`) + fp32 denominator
  v16bf bF;
  float denom = 0.f;
#pragma unroll
  for (int j = 0; j < 8; ++j) {
    const int de = hs * 16 + 2 * j;
    const float qe = Qp[(size_t)(de * NH + h) * L + col];
    const float qo = Qp[(size_t)((de + 1) * NH + h) * L + col];
    const float2 ks2 = *reinterpret_cast<const float2*>(ksv + de);
    denom += qe * ks2.x + qo * ks2.y;
    bF[2 * j]     = (__bf16)qe;
    bF[2 * j + 1] = (__bf16)qo;
  }
  denom += __shfl_xor(denom, 16);              // combine lane halves (wave32)
  const float zscale = (float)L / (denom + EPSV);

  v8f acc0 = wmma_bf16(aF[0], bF, zero_v8f());
  v8f acc1 = wmma_bf16(aF[1], bF, zero_v8f());

  float* xb = X2 + (size_t)b * DM * L;
#pragma unroll
  for (int r = 0; r < 8; ++r) {
    const int q0 = r + hs * 8;                 // C layout: m = r + 8*hs
    xb[(size_t)(q0 * NH + h) * L + col]        = acc0[r] * zscale;
    xb[(size_t)((q0 + 16) * NH + h) * L + col] = acc1[r] * zscale;
  }
}

// ---------------------------------------------------------------------------
extern "C" void kernel_launch(void* const* d_in, const int* in_sizes, int n_in,
                              void* d_out, int out_size, void* d_ws, size_t ws_size,
                              hipStream_t stream) {
  (void)n_in; (void)out_size; (void)ws_size;
  const float* query = (const float*)d_in[0];
  const float* key   = (const float*)d_in[1];
  const float* value = (const float*)d_in[2];
  const float* Wq = (const float*)d_in[3];
  const float* bq = (const float*)d_in[4];
  const float* Wk = (const float*)d_in[5];
  const float* bk = (const float*)d_in[6];
  const float* Wv = (const float*)d_in[7];
  const float* bv = (const float*)d_in[8];
  const float* Wm = (const float*)d_in[9];
  const float* bm = (const float*)d_in[10];

  const int L = in_sizes[0] / (BATCH * DM);    // 4096
  const size_t planeBytes = (size_t)BATCH * DM * (size_t)L * sizeof(float);

  char* ws = (char*)d_ws;
  float* Qbuf = (float*)(ws);
  float* Kbuf = (float*)(ws + planeBytes);
  float* Vbuf = (float*)(ws + 2 * planeBytes);
  float* X2   = (float*)(ws + 3 * planeBytes);
  float* KVb  = (float*)(ws + 4 * planeBytes);           // 64 * 32*32 fp32
  float* KSb  = KVb + (size_t)BATCH * NH * 1024;         // 64 * 32 fp32

  dim3 gP(L / 128, DM / 128, BATCH);
  // Q = elu(Wq x + bq) + 1 ; K = elu(Wk x + bk) + 1 ; V = (Wv x + bv)/L
  proj_gemm_kernel<1><<<gP, 256, 0, stream>>>(query, Wq, bq, Qbuf, L, 1.f);
  proj_gemm_kernel<1><<<gP, 256, 0, stream>>>(key,   Wk, bk, Kbuf, L, 1.f);
  proj_gemm_kernel<2><<<gP, 256, 0, stream>>>(value, Wv, bv, Vbuf, L, 1.f / (float)L);
  // KV and ksum per (b,h)
  kv_ksum_kernel<<<BATCH * NH, 256, 0, stream>>>(Kbuf, Vbuf, KVb, KSb, L);
  // normalized attention output, heads re-merged into channels
  attn_out_kernel<<<dim3(L / 128, NH, BATCH), 256, 0, stream>>>(Qbuf, KVb, KSb, X2, L);
  // final projection straight into d_out
  proj_gemm_kernel<0><<<gP, 256, 0, stream>>>(X2, Wm, bm, (float*)d_out, L, 1.f);
}
